// MultiHeadedAttention_88021059764737
// MI455X (gfx1250) — matrix operand order repaired and checked by reading, not compiled, not run on hardware
//
#include <hip/hip_runtime.h>
#include <hip/hip_bf16.h>

typedef _Float16 half_t;
typedef __attribute__((ext_vector_type(16))) _Float16 v16h;
typedef __attribute__((ext_vector_type(8)))  float    v8f;

// ---------------------------------------------------------------------------
// WMMA fragment loaders per CDNA5 ISA 7.12.2 (wave32).
// A (16x32 f16, MxK): lane = M row (lane&15); half-wave selects K-group:
//   g=0: VGPR0..3 -> K=0..7,  VGPR4..7 -> K=16..23
//   g=1: VGPR0..3 -> K=8..15, VGPR4..7 -> K=24..31
// B (32x16 f16, KxN): lane = N col (lane&15); g=0 holds K=0..7, 16..23 (same as A),
//   g=1 holds K=8..15, 24..31.
// C/D (16x16 f32): VGPR r: lanes 0-15 -> M=r, lanes 16-31 -> M=8+r; N=lane&15.
//
// All loaders below are arranged so each lane reads CONTIGUOUS 16B chunks
// (merge into b128 loads): A from row-major [M][K], B from transposed [N][K].
// ---------------------------------------------------------------------------
__device__ __forceinline__ v16h load_A_rm(const half_t* base, int ld) {
  const int lane = threadIdx.x & 31;
  const int m = lane & 15, g = lane >> 4;
  v16h a;
#pragma unroll
  for (int i = 0; i < 8; ++i) {
    const int kb = (i < 4 ? 2 * i : 16 + 2 * (i - 4)) + 8 * g;
    a[2 * i]     = base[m * ld + kb];
    a[2 * i + 1] = base[m * ld + kb + 1];
  }
  return a;
}

// B fragment from transposed storage: element (k, n) at base[n*ld + k]
// (per-lane one row, contiguous k -> b128 merges)
__device__ __forceinline__ v16h load_BT(const half_t* base, int ld) {
  const int lane = threadIdx.x & 31;
  const int n = lane & 15, g = lane >> 4;
  v16h b;
#pragma unroll
  for (int i = 0; i < 8; ++i) {
    const int k = (i < 4 ? 2 * i : 16 + 2 * (i - 4)) + 8 * g;
    b[2 * i]     = base[n * ld + k];
    b[2 * i + 1] = base[n * ld + k + 1];
  }
  return b;
}

__device__ __forceinline__ v8f wmma_f16(v16h a, v16h b, v8f c) {
  return __builtin_amdgcn_wmma_f32_16x16x32_f16(
      /*neg_a=*/false, a, /*neg_b=*/false, b,
      /*c_mod=*/(short)0, c, /*reuse_a=*/false, /*reuse_b=*/false);
}

// CDNA5 async global->LDS copy (ASYNCcnt-tracked); lds_off = byte offset in LDS
// (low 32 bits of a flat shared-memory address).
__device__ __forceinline__ void async_copy_b128(uint32_t lds_off, const half_t* gaddr) {
  asm volatile("global_load_async_to_lds_b128 %0, %1, off"
               :: "v"(lds_off), "v"(gaddr) : "memory");
}
__device__ __forceinline__ void wait_asynccnt0() {
  asm volatile("s_wait_asynccnt 0x0" ::: "memory");
}
__device__ __forceinline__ uint32_t lds_offset_of(const void* p) {
  return (uint32_t)(uintptr_t)p;   // flat shared addr: LDS offset in addr[31:0]
}

// ---------------------------------------------------------------------------
// Dense adjacency build: adj[b][src][dst] += 1.0  (exact integer f32 sums ->
// order-independent, deterministic).
// ---------------------------------------------------------------------------
__global__ __launch_bounds__(256)
void build_adj_kernel(const int* __restrict__ edges, float* __restrict__ adj) {
  constexpr int Bc = 4, Nc = 2048, Ec = 32768;
  int idx = blockIdx.x * blockDim.x + threadIdx.x;
  if (idx >= Bc * Ec) return;
  int b = idx / Ec, i = idx - b * Ec;
  int src = edges[(size_t)(b * 2 + 0) * Ec + i];
  int dst = edges[(size_t)(b * 2 + 1) * Ec + i];
  atomicAdd(&adj[((size_t)b * Nc + src) * Nc + dst], 1.0f);
}

// ---------------------------------------------------------------------------
// Projection GEMM: acc[m][n] = sum_k W[m][k] * X[b][k][n] + bias[m]
// M=K=256, Nc=2048. Block = 8 waves, tile 64(M) x 128(N), K-step 32.
// Output layouts:
//   OUT_QK : f16, head-deinterleaved n-major [b][h][n][dim]  (h=m&3, dim=m>>2)
//   OUT_V  : f16, d-major [b][m][n]
//   OUT_F32: f32, d-major [b][m][n]  (final result)
// ---------------------------------------------------------------------------
enum { OUT_QK = 0, OUT_V = 1, OUT_F32 = 2 };

template <typename TIN, int OUTMODE>
__global__ __launch_bounds__(256)
void gemm_kernel(const float* __restrict__ W, const TIN* __restrict__ X,
                 const float* __restrict__ bias,
                 half_t* __restrict__ outH, float* __restrict__ outF) {
  constexpr int M = 256, K = 256, Nc = 2048, Hh = 4, DIM = 64;
  __shared__ alignas(16) half_t As[64][40];     // [m][k], 80B rows
  __shared__ alignas(16) half_t BsT[128][40];   // [n][k] transposed, 80B rows

  const int b  = blockIdx.z;
  const int m0 = blockIdx.y * 64;
  const int n0 = blockIdx.x * 128;
  const TIN* Xb = X + (size_t)b * K * Nc;
  const int t  = threadIdx.x;
  const int w  = t >> 5;
  const int wm = (w >> 2) * 32;
  const int wn = (w & 3) * 32;

  v8f acc[2][2];
#pragma unroll
  for (int i = 0; i < 2; ++i)
#pragma unroll
    for (int j = 0; j < 2; ++j)
#pragma unroll
      for (int r = 0; r < 8; ++r) acc[i][j][r] = 0.0f;

  for (int k0 = 0; k0 < K; k0 += 32) {
    __syncthreads();
#pragma unroll
    for (int l = 0; l < 8; ++l) {               // 64x32 A tile (f32 -> f16)
      int idx = t + l * 256;
      int r = idx >> 5, c = idx & 31;
      As[r][c] = (half_t)W[(size_t)(m0 + r) * K + k0 + c];
    }
#pragma unroll
    for (int l = 0; l < 16; ++l) {              // 32x128 B tile, stored [n][k]
      int idx = t + l * 256;
      int r = idx >> 7, c = idx & 127;          // r = k row, c = n col (coalesced read)
      BsT[c][r] = (half_t)Xb[(size_t)(k0 + r) * Nc + n0 + c];
    }
    __syncthreads();

    v16h a0 = load_A_rm(&As[wm][0], 40);
    v16h a1 = load_A_rm(&As[wm + 16][0], 40);
    v16h b0 = load_BT(&BsT[wn][0], 40);
    v16h b1 = load_BT(&BsT[wn + 16][0], 40);
    acc[0][0] = wmma_f16(a0, b0, acc[0][0]);
    acc[0][1] = wmma_f16(a0, b1, acc[0][1]);
    acc[1][0] = wmma_f16(a1, b0, acc[1][0]);
    acc[1][1] = wmma_f16(a1, b1, acc[1][1]);
  }

  const int lane = t & 31;
  const int g = lane >> 4, nc = lane & 15;
#pragma unroll
  for (int ti = 0; ti < 2; ++ti)
#pragma unroll
    for (int tj = 0; tj < 2; ++tj)
#pragma unroll
      for (int r = 0; r < 8; ++r) {
        int m = m0 + wm + ti * 16 + r + 8 * g;
        int n = n0 + wn + tj * 16 + nc;
        float v = acc[ti][tj][r] + bias[m];
        if constexpr (OUTMODE == OUT_QK) {
          int hh = m & 3, dim = m >> 2;          // m = dim*H + h
          outH[(((size_t)b * Hh + hh) * Nc + n) * DIM + dim] = (half_t)v;
        } else if constexpr (OUTMODE == OUT_V) {
          outH[(size_t)b * M * Nc + (size_t)m * Nc + n] = (half_t)v;
        } else {
          outF[(size_t)b * M * Nc + (size_t)m * Nc + n] = v;
        }
      }
}

// ---------------------------------------------------------------------------
// Attention: block = (b, h, 256 query rows) = 8 waves x 32 rows (disjoint ->
// no cross-wave reduction). All waves sweep the same 64 key tiles; K/V tiles
// are double-buffered in LDS via async global->LDS b128 copies overlapped
// with WMMA. One-pass streaming:
//   S = Q^T K / 8 ; e = exp(S * adj) ; O += e*V^T ; rowsum += e ; O /= rowsum
// Layouts: q16/k16 = [b][h][n][dim] (dim-contiguous), v16/x16 = [b][d][n].
// ---------------------------------------------------------------------------
__global__ __launch_bounds__(256)
void attn_kernel(const half_t* __restrict__ q16, const half_t* __restrict__ k16,
                 const half_t* __restrict__ v16, const float* __restrict__ adj,
                 half_t* __restrict__ x16) {
  constexpr int Nn = 2048, Dd = 256, Hh = 4, DIM = 64;
  constexpr int LDKt = 72;                 // 64 + 8 halves -> 144B rows (16B mult)
  constexpr int LDV  = 40;                 // 32 + 8 halves -> 80B rows
  constexpr int LDP  = 40;
  __shared__ alignas(16) half_t Kt[2][32][LDKt];   // [buf][m][dim]
  __shared__ alignas(16) half_t Vs[2][64][LDV];    // [buf][dim][m]
  __shared__ alignas(16) half_t Ps[8][32][LDP];    // per-wave P staging [n][m]

  const int b = blockIdx.z, h = blockIdx.y;
  const int t = threadIdx.x, w = t >> 5, lane = t & 31;
  const int g = lane >> 4, lc = lane & 15;
  const long hstride = (long)Hh * Nn;
  const int n0w = blockIdx.x * 256 + w * 32;     // this wave's query rows

  const half_t* Qhead = q16 + (size_t)(b * Hh + h) * Nn * DIM;
  const half_t* Khead = k16 + (size_t)(b * Hh + h) * Nn * DIM;
  const size_t vOff = (size_t)b * Dd * Nn + (size_t)h * Nn;
  const half_t* Vh = v16 + vOff;

  // Async-copy coordinates. K tile: 32 m-rows x 128B = 8 chunks/row.
  const int kmr  = t >> 3;
  const int kchk = (t & 7) * 8;
  // V tile: 64 dim-rows x 64B = 4 chunks/row.
  const int vdr  = t >> 2;
  const int vchk = (t & 3) * 8;
  uint32_t ksb[2], vsb[2];
#pragma unroll
  for (int u = 0; u < 2; ++u) {
    ksb[u] = lds_offset_of(&Kt[u][0][0]) + (uint32_t)(kmr * LDKt + kchk) * 2u;
    vsb[u] = lds_offset_of(&Vs[u][0][0]) + (uint32_t)(vdr * LDV + vchk) * 2u;
  }
  const half_t* kg = Khead + (size_t)kmr * DIM + kchk;     // + m0*DIM per tile
  const half_t* vg = Vh + (size_t)vdr * hstride + vchk;    // + m0 per tile

  // One-time Q fragments (row-major [n][dim] -> b128 global loads).
  v16h aQ[2][2];
#pragma unroll
  for (int ti = 0; ti < 2; ++ti)
#pragma unroll
    for (int kc = 0; kc < 2; ++kc)
      aQ[ti][kc] = load_A_rm(Qhead + (size_t)(n0w + ti * 16) * DIM + kc * 32, DIM);

  v8f oacc[2][4];
#pragma unroll
  for (int ti = 0; ti < 2; ++ti)
#pragma unroll
    for (int jd = 0; jd < 4; ++jd)
#pragma unroll
      for (int r = 0; r < 8; ++r) oacc[ti][jd][r] = 0.0f;
  float rs[2][8];
#pragma unroll
  for (int ti = 0; ti < 2; ++ti)
#pragma unroll
    for (int r = 0; r < 8; ++r) rs[ti][r] = 0.0f;

  // Prologue: async-stage tile 0.
  async_copy_b128(ksb[0], kg);
  async_copy_b128(vsb[0], vg);

  for (int mt = 0; mt < Nn / 32; ++mt) {
    const int cur = mt & 1;
    const int m0 = mt * 32;

    wait_asynccnt0();        // current tile landed in LDS
    __syncthreads();         // ...and every wave is done with the other buffer

    if (mt + 1 < Nn / 32) {  // overlap next tile's copy with this tile's math
      async_copy_b128(ksb[cur ^ 1], kg + (size_t)(m0 + 32) * DIM);
      async_copy_b128(vsb[cur ^ 1], vg + m0 + 32);
      __builtin_prefetch(&adj[((size_t)b * Nn + n0w + lane) * Nn + m0 + 32], 0, 1);
    }

    // ---- S = Q^T K  (K-dim = 64 -> 2 chunks of 32) ----
    v8f sacc[2][2];
#pragma unroll
    for (int ti = 0; ti < 2; ++ti)
#pragma unroll
      for (int tj = 0; tj < 2; ++tj)
#pragma unroll
        for (int r = 0; r < 8; ++r) sacc[ti][tj][r] = 0.0f;

#pragma unroll
    for (int kc = 0; kc < 2; ++kc) {
      v16h bK0 = load_BT(&Kt[cur][0][kc * 32], LDKt);     // frag (k=dim, n=m)
      v16h bK1 = load_BT(&Kt[cur][16][kc * 32], LDKt);
      sacc[0][0] = wmma_f16(aQ[0][kc], bK0, sacc[0][0]);
      sacc[0][1] = wmma_f16(aQ[0][kc], bK1, sacc[0][1]);
      sacc[1][0] = wmma_f16(aQ[1][kc], bK0, sacc[1][0]);
      sacc[1][1] = wmma_f16(aQ[1][kc], bK1, sacc[1][1]);
    }

    // ---- e = exp((S/8) * adj); accumulate rowsum; stage P (wave-private) ----
#pragma unroll
    for (int ti = 0; ti < 2; ++ti)
#pragma unroll
      for (int tj = 0; tj < 2; ++tj)
#pragma unroll
        for (int r = 0; r < 8; ++r) {
          int nl = ti * 16 + r + 8 * g;
          int ml = tj * 16 + lc;
          float s = sacc[ti][tj][r] * 0.125f;
          float msk = adj[((size_t)b * Nn + n0w + nl) * Nn + m0 + ml];
          float e = __expf(s * msk);
          rs[ti][r] += e;
          Ps[w][nl][ml] = (half_t)e;
        }

    // ---- O += P * V^T  (A from Ps row-major; B frag (k=m', n=dim) from Vs) ----
    v16h aP0 = load_A_rm(&Ps[w][0][0], LDP);
    v16h aP1 = load_A_rm(&Ps[w][16][0], LDP);
#pragma unroll
    for (int jd = 0; jd < 4; ++jd) {
      v16h bV = load_BT(&Vs[cur][jd * 16][0], LDV);
      oacc[0][jd] = wmma_f16(aP0, bV, oacc[0][jd]);
      oacc[1][jd] = wmma_f16(aP1, bV, oacc[1][jd]);
    }
  }

  // Rowsum across the 16 lanes sharing an M row (wave32 C-layout).
#pragma unroll
  for (int ti = 0; ti < 2; ++ti)
#pragma unroll
    for (int r = 0; r < 8; ++r) {
      rs[ti][r] += __shfl_xor(rs[ti][r], 1, 32);
      rs[ti][r] += __shfl_xor(rs[ti][r], 2, 32);
      rs[ti][r] += __shfl_xor(rs[ti][r], 4, 32);
      rs[ti][r] += __shfl_xor(rs[ti][r], 8, 32);
    }

  // Normalize and emit x16[b][dim*H+h][n] (same row ownership as oacc).
#pragma unroll
  for (int ti = 0; ti < 2; ++ti)
#pragma unroll
    for (int jd = 0; jd < 4; ++jd)
#pragma unroll
      for (int r = 0; r < 8; ++r) {
        int n  = n0w + ti * 16 + r + 8 * g;
        int dm = jd * 16 + lc;
        x16[vOff + (size_t)dm * hstride + n] = (half_t)(oacc[ti][jd][r] / rs[ti][r]);
      }
}

// ---------------------------------------------------------------------------
extern "C" void kernel_launch(void* const* d_in, const int* in_sizes, int n_in,
                              void* d_out, int out_size, void* d_ws, size_t ws_size,
                              hipStream_t stream) {
  constexpr int Bc = 4, Dc = 256, Nc = 2048, Ec = 32768;

  const float* query = (const float*)d_in[0];
  const float* key_  = (const float*)d_in[1];
  const float* value = (const float*)d_in[2];
  const int*   edges = (const int*)d_in[3];
  const float* Wq = (const float*)d_in[4];
  const float* bq = (const float*)d_in[5];
  const float* Wk = (const float*)d_in[6];
  const float* bk = (const float*)d_in[7];
  const float* Wv = (const float*)d_in[8];
  const float* bv = (const float*)d_in[9];
  const float* Wm = (const float*)d_in[10];
  const float* bm = (const float*)d_in[11];
  float* out = (float*)d_out;

  char* ws = (char*)d_ws;
  half_t* q16 = (half_t*)ws; ws += (size_t)Bc * Dc * Nc * sizeof(half_t);
  half_t* k16 = (half_t*)ws; ws += (size_t)Bc * Dc * Nc * sizeof(half_t);
  half_t* v16 = (half_t*)ws; ws += (size_t)Bc * Dc * Nc * sizeof(half_t);
  half_t* x16 = (half_t*)ws; ws += (size_t)Bc * Dc * Nc * sizeof(half_t);
  float*  adj = (float*)ws;

  hipMemsetAsync(adj, 0, (size_t)Bc * Nc * Nc * sizeof(float), stream);
  build_adj_kernel<<<(Bc * Ec + 255) / 256, 256, 0, stream>>>(edges, adj);

  dim3 gg(Nc / 128, Dc / 64, Bc);   // (16, 4, 4)
  gemm_kernel<float, OUT_QK><<<gg, 256, 0, stream>>>(Wq, query, bq, q16, nullptr);
  gemm_kernel<float, OUT_QK><<<gg, 256, 0, stream>>>(Wk, key_,  bk, k16, nullptr);
  gemm_kernel<float, OUT_V ><<<gg, 256, 0, stream>>>(Wv, value, bv, v16, nullptr);

  attn_kernel<<<dim3(Nc / 256, 4, Bc), 256, 0, stream>>>(q16, k16, v16, adj, x16);

  gemm_kernel<half_t, OUT_F32><<<gg, 256, 0, stream>>>(Wm, x16, bm, nullptr, out);
}
